// MultiHeadAttention_24996709662856
// MI455X (gfx1250) — matrix operand order repaired and checked by reading, not compiled, not run on hardware
//
#include <hip/hip_runtime.h>

typedef __attribute__((ext_vector_type(16))) __bf16 v16bf;
typedef __attribute__((ext_vector_type(8)))  float  v8f;
typedef __attribute__((ext_vector_type(4)))  float  f32x4;
typedef __attribute__((ext_vector_type(4)))  int    v4i;

namespace {
constexpr int Bb    = 4;
constexpr int Ss    = 2048;
constexpr int Dd    = 1024;
constexpr int Hh    = 16;
constexpr int Mrows = Bb * Ss;  // 8192
}

__device__ __forceinline__ v8f wmma_bf16(const v16bf& a, const v16bf& b, const v8f& c) {
  // D = A(16x32 bf16) x B(32x16 bf16) + C(16x16 f32)
  return __builtin_amdgcn_wmma_f32_16x16x32_bf16(false, a, false, b, (short)0, c, false, false);
}

__device__ __forceinline__ void load16(v16bf& dst, const __bf16* p0, const __bf16* p1) {
  *(f32x4*)&dst       = *(const f32x4*)p0;
  *((f32x4*)&dst + 1) = *(const f32x4*)p1;
}

// 16B global -> LDS copy. Prefer CDNA5 async-to-LDS (ASYNCcnt path, no VGPR
// round trip); fall back to a plain load/store pair if the builtin is absent.
// Builtin signature (per hipcc diagnostic): (v4i AS1*, v4i AS3*, imm offset, imm cpol).
__device__ __forceinline__ void async_b128(void* lds, const void* g) {
#if __has_builtin(__builtin_amdgcn_global_load_async_to_lds_b128)
  __builtin_amdgcn_global_load_async_to_lds_b128(
      (__attribute__((address_space(1))) v4i*)g,
      (__attribute__((address_space(3))) v4i*)lds, 0, 0);
#else
  *(f32x4*)lds = *(const f32x4*)g;
#endif
}

__device__ __forceinline__ void wait_async() {
#if __has_builtin(__builtin_amdgcn_s_wait_asynccnt)
  __builtin_amdgcn_s_wait_asynccnt(0);
#else
  asm volatile("s_wait_asynccnt 0x0" ::: "memory");
#endif
}

__global__ void __launch_bounds__(256)
cvt_f32_to_bf16(const float* __restrict__ in, __bf16* __restrict__ out, int n4) {
  int i = blockIdx.x * 256 + threadIdx.x;
  if (i < n4) {
    f32x4 v = ((const f32x4*)in)[i];
    out[4 * i + 0] = (__bf16)v[0];
    out[4 * i + 1] = (__bf16)v[1];
    out[4 * i + 2] = (__bf16)v[2];
    out[4 * i + 3] = (__bf16)v[3];
  }
}

// C[M,1024] = A[M,1024] @ W[1024,1024]^T   (both A and W rows are K-contiguous)
// Block: 256 threads = 8 waves; tile 128(M) x 128(N); K tile 64, double-buffered
// via async-to-LDS. Wave (wm,wn) grid 4x2; each wave: 32x64 = 2x4 WMMA tiles.
template <bool OUT_F32>
__global__ void __launch_bounds__(256)
gemm_xwt(const __bf16* __restrict__ A, const __bf16* __restrict__ W, void* __restrict__ Cout) {
  __shared__ __bf16 As[2][128 * 64];  // 16KB x2
  __shared__ __bf16 Ws[2][128 * 64];  // 16KB x2
  const int tid  = threadIdx.x;
  const int wave = tid >> 5;
  const int lane = tid & 31;
  const int lg   = lane >> 4;   // lane group (WMMA K-half selector)
  const int lr   = lane & 15;   // row-within-16
  const int bm   = blockIdx.x * 128;
  const int bn   = blockIdx.y * 128;
  const int wm   = (wave >> 1) * 32;
  const int wn   = (wave & 1) * 64;

  // Stage one 128x64 bf16 tile of A and W: 1024 16B chunks each, 4 per thread.
  auto stage = [&](int buf, int k0) {
#pragma unroll
    for (int u = 0; u < 4; ++u) {
      int c = tid + u * 256;
      int row = c >> 3, cc = c & 7;
      async_b128(&As[buf][c * 8], &A[(size_t)(bm + row) * Dd + k0 + cc * 8]);
      async_b128(&Ws[buf][c * 8], &W[(size_t)(bn + row) * Dd + k0 + cc * 8]);
    }
  };

  const v8f zero = {};
  v8f acc[2][4];
#pragma unroll
  for (int mt = 0; mt < 2; ++mt)
#pragma unroll
    for (int nt = 0; nt < 4; ++nt) acc[mt][nt] = zero;

  stage(0, 0);
  wait_async();
  __syncthreads();

  int buf = 0;
  for (int k0 = 0; k0 < Dd; k0 += 64, buf ^= 1) {
    if (k0 + 64 < Dd) stage(buf ^ 1, k0 + 64);  // prefetch next tile (async)

#pragma unroll
    for (int ks = 0; ks < 2; ++ks) {
      v16bf af[2], bfr[4];
#pragma unroll
      for (int mt = 0; mt < 2; ++mt) {  // A-frag: lanes<16 K{0..7,16..23}, lanes>=16 K{8..15,24..31}
        const __bf16* p = &As[buf][(wm + mt * 16 + lr) * 64 + ks * 32 + lg * 8];
        load16(af[mt], p, p + 16);
      }
#pragma unroll
      for (int nt = 0; nt < 4; ++nt) {  // B-frag: lane holds N=lr, lanes<16 K{0..7,16..23}, lanes>=16 K{8..15,24..31}
        const __bf16* p = &Ws[buf][(wn + nt * 16 + lr) * 64 + ks * 32 + lg * 8];
        load16(bfr[nt], p, p + 16);
      }
#pragma unroll
      for (int mt = 0; mt < 2; ++mt)
#pragma unroll
        for (int nt = 0; nt < 4; ++nt) acc[mt][nt] = wmma_bf16(af[mt], bfr[nt], acc[mt][nt]);
    }

    wait_async();     // next tile landed in LDS
    __syncthreads();  // everyone done reading buf / writing buf^1
  }

#pragma unroll
  for (int mt = 0; mt < 2; ++mt)
#pragma unroll
    for (int nt = 0; nt < 4; ++nt)
#pragma unroll
      for (int r = 0; r < 8; ++r) {  // D layout: vgpr r holds M = r + 8*lg
        size_t row = (size_t)(bm + wm + mt * 16 + r + lg * 8);
        size_t col = (size_t)(bn + wn + nt * 16 + lr);
        float v = acc[mt][nt][r];
        if (OUT_F32) ((float*)Cout)[row * Dd + col] = v;
        else         ((__bf16*)Cout)[row * Dd + col] = (__bf16)v;
      }
}

// Flash-style attention. Grid: (S/64, H, B). Block: 128 threads = 4 waves.
// Each block: 64 query rows of one (b,h); each wave owns 16 rows.
// Projections stored [B*S, 1024]; head h occupies columns h*64..h*64+63.
__global__ void __launch_bounds__(128)
attn_kernel(const __bf16* __restrict__ Qp, const __bf16* __restrict__ Kp,
            const __bf16* __restrict__ Vp, __bf16* __restrict__ Ctx) {
  __shared__ float  Sld[64 * 64];   // raw scores (f32)
  __shared__ __bf16 Pld[64 * 64];   // exp probabilities (bf16)
  __shared__ __bf16 Vt[64 * 64];    // V tile transposed: [dh][key]
  __shared__ float  row_m[64], row_l[64], row_scale[64];

  const int q0   = blockIdx.x * 64;
  const int h    = blockIdx.y;
  const int b    = blockIdx.z;
  const size_t rowbase = (size_t)b * Ss * Dd;
  const int hcol = h * 64;

  const int tid  = threadIdx.x;
  const int wave = tid >> 5;
  const int lane = tid & 31;
  const int lg   = lane >> 4;
  const int lr   = lane & 15;
  const int wrow = wave * 16;  // this wave's q rows within the 64-row block

  if (tid < 64) { row_m[tid] = -__builtin_inff(); row_l[tid] = 0.0f; }

  // Q fragments for this wave's 16 rows, dh split into two K-steps of 32 — kept live all kernel.
  v16bf qfr[2];
  {
    const __bf16* qrow = Qp + rowbase + (size_t)(q0 + wrow + lr) * Dd + hcol;
#pragma unroll
    for (int ks = 0; ks < 2; ++ks) {
      const __bf16* p = qrow + ks * 32 + lg * 8;
      load16(qfr[ks], p, p + 16);
    }
  }

  const v8f zero = {};
  v8f o[4];
#pragma unroll
  for (int nt = 0; nt < 4; ++nt) o[nt] = zero;

  for (int kb = 0; kb < Ss; kb += 64) {
    __syncthreads();  // previous iteration's Vt/Pld consumers done

    // Stage V tile transposed into LDS. Batch the 4 global loads per thread
    // first (keep them all in flight), then scatter-store the transpose.
    f32x4 vtmp[4];
#pragma unroll
    for (int u = 0; u < 4; ++u) {
      int c = tid + u * 128;          // c in [0,512): key = c>>3, dh0 = (c&7)*8
      int key = c >> 3, dh0 = (c & 7) * 8;
      vtmp[u] = *(const f32x4*)&Vp[rowbase + (size_t)(kb + key) * Dd + hcol + dh0];
    }
#pragma unroll
    for (int u = 0; u < 4; ++u) {
      int c = tid + u * 128;
      int key = c >> 3, dh0 = (c & 7) * 8;
      const __bf16* t = (const __bf16*)&vtmp[u];
#pragma unroll
      for (int j = 0; j < 8; ++j) Vt[(dh0 + j) * 64 + key] = t[j];
    }

    // Scores: this wave's 16 rows x 64 keys = 4 tiles, 2 K-steps over dh.
    v8f sc[4];
#pragma unroll
    for (int nt = 0; nt < 4; ++nt) sc[nt] = zero;
#pragma unroll
    for (int ks = 0; ks < 2; ++ks) {
#pragma unroll
      for (int nt = 0; nt < 4; ++nt) {
        v16bf kfr;  // B-frag = K^T: lane holds key = kb+nt*16+lr, dh {lg*8..+7, 16+lg*8..+7}
        const __bf16* p = Kp + rowbase + (size_t)(kb + nt * 16 + lr) * Dd + hcol + ks * 32 + lg * 8;
        load16(kfr, p, p + 16);
        sc[nt] = wmma_bf16(qfr[ks], kfr, sc[nt]);
      }
    }
#pragma unroll
    for (int nt = 0; nt < 4; ++nt)
#pragma unroll
      for (int r = 0; r < 8; ++r)
        Sld[(wrow + r + lg * 8) * 64 + nt * 16 + lr] = sc[nt][r] * 0.125f;  // 1/sqrt(64)
    __syncthreads();

    // Online softmax: one thread per query row.
    if (tid < 64) {
      float mx = row_m[tid], tm = -__builtin_inff();
      for (int j = 0; j < 64; ++j) tm = fmaxf(tm, Sld[tid * 64 + j]);
      float nm  = fmaxf(mx, tm);
      float scl = __expf(mx - nm);  // exp(-inf)=0 on first tile
      float sum = 0.0f;
      for (int j = 0; j < 64; ++j) {
        float p = __expf(Sld[tid * 64 + j] - nm);
        sum += p;
        Pld[tid * 64 + j] = (__bf16)p;
      }
      row_l[tid]     = row_l[tid] * scl + sum;
      row_m[tid]     = nm;
      row_scale[tid] = scl;
    }
    __syncthreads();

    // Rescale running O accumulators by per-row factor.
    float scl[8];
#pragma unroll
    for (int r = 0; r < 8; ++r) scl[r] = row_scale[wrow + r + lg * 8];
#pragma unroll
    for (int nt = 0; nt < 4; ++nt)
#pragma unroll
      for (int r = 0; r < 8; ++r) o[nt][r] *= scl[r];

    // O += P @ V : K-dim = 64 keys, 2 K-steps.
#pragma unroll
    for (int ks = 0; ks < 2; ++ks) {
      v16bf pfr;
      const __bf16* pp = &Pld[(wrow + lr) * 64 + ks * 32 + lg * 8];
      load16(pfr, pp, pp + 16);
#pragma unroll
      for (int nt = 0; nt < 4; ++nt) {
        v16bf vfr;  // lane holds dh = nt*16+lr, keys {lg*8..+7, 16+lg*8..+7} (from transposed Vt)
        const __bf16* vp = &Vt[(nt * 16 + lr) * 64 + ks * 32 + lg * 8];
        load16(vfr, vp, vp + 16);
        o[nt] = wmma_bf16(pfr, vfr, o[nt]);
      }
    }
  }

  __syncthreads();
  float invl[8];
#pragma unroll
  for (int r = 0; r < 8; ++r) invl[r] = 1.0f / row_l[wrow + r + lg * 8];
#pragma unroll
  for (int nt = 0; nt < 4; ++nt)
#pragma unroll
    for (int r = 0; r < 8; ++r) {
      size_t row = (size_t)(q0 + wrow + r + lg * 8);
      size_t col = (size_t)(hcol + nt * 16 + lr);
      Ctx[rowbase + row * Dd + col] = (__bf16)(o[nt][r] * invl[r]);
    }
}

extern "C" void kernel_launch(void* const* d_in, const int* in_sizes, int n_in,
                              void* d_out, int out_size, void* d_ws, size_t ws_size,
                              hipStream_t stream) {
  const float* q   = (const float*)d_in[0];
  const float* k   = (const float*)d_in[1];
  const float* v   = (const float*)d_in[2];
  const float* w_q = (const float*)d_in[3];
  const float* w_k = (const float*)d_in[4];
  const float* w_v = (const float*)d_in[5];
  const float* w_o = (const float*)d_in[6];
  float* out = (float*)d_out;

  const size_t nQKV = (size_t)Mrows * Dd;  // 8,388,608
  const size_t nW   = (size_t)Dd * Dd;     // 1,048,576

  char* ws = (char*)d_ws;
  size_t off = 0;
  auto take = [&](size_t nelem) -> __bf16* {
    __bf16* p = (__bf16*)(ws + off);
    off += (nelem * 2 + 255) & ~(size_t)255;
    return p;
  };
  __bf16* qb = take(nQKV);
  __bf16* kb = take(nQKV);
  __bf16* vb = take(nQKV);
  __bf16* wq = take(nW);
  __bf16* wk = take(nW);
  __bf16* wv = take(nW);
  __bf16* wo = take(nW);
  __bf16* Qp = take(nQKV);
  __bf16* Kp = take(nQKV);
  __bf16* Vp = take(nQKV);
  __bf16* Cx = qb;  // qb dead after its projection; reuse as attention context

  const int n4qkv = (int)(nQKV / 4), n4w = (int)(nW / 4);
  cvt_f32_to_bf16<<<(n4qkv + 255) / 256, 256, 0, stream>>>(q, qb, n4qkv);
  cvt_f32_to_bf16<<<(n4qkv + 255) / 256, 256, 0, stream>>>(k, kb, n4qkv);
  cvt_f32_to_bf16<<<(n4qkv + 255) / 256, 256, 0, stream>>>(v, vb, n4qkv);
  cvt_f32_to_bf16<<<(n4w + 255) / 256, 256, 0, stream>>>(w_q, wq, n4w);
  cvt_f32_to_bf16<<<(n4w + 255) / 256, 256, 0, stream>>>(w_k, wk, n4w);
  cvt_f32_to_bf16<<<(n4w + 255) / 256, 256, 0, stream>>>(w_v, wv, n4w);
  cvt_f32_to_bf16<<<(n4w + 255) / 256, 256, 0, stream>>>(w_o, wo, n4w);

  dim3 gg(Mrows / 128, Dd / 128), gb(256);
  gemm_xwt<false><<<gg, gb, 0, stream>>>(qb, wq, Qp);
  gemm_xwt<false><<<gg, gb, 0, stream>>>(kb, wk, Kp);
  gemm_xwt<false><<<gg, gb, 0, stream>>>(vb, wv, Vp);

  attn_kernel<<<dim3(Ss / 64, Hh, Bb), 128, 0, stream>>>(Qp, Kp, Vp, Cx);

  gemm_xwt<true><<<gg, gb, 0, stream>>>(Cx, wo, (void*)out);
}
